// CausalSelfAttention_38216619000057
// MI455X (gfx1250) — matrix operand order repaired and checked by reading, not compiled, not run on hardware
//
#include <hip/hip_runtime.h>
#include <hip/hip_bf16.h>
#include <cstddef>

typedef __attribute__((ext_vector_type(16))) __bf16 v16bf;
typedef __attribute__((ext_vector_type(8)))  __bf16 v8bf;
typedef __attribute__((ext_vector_type(8)))  float  v8f;

__device__ __forceinline__ unsigned short f32_to_bf16(float f) {
    unsigned u = __float_as_uint(f);
    u += 0x7FFFu + ((u >> 16) & 1u);   // round to nearest even
    return (unsigned short)(u >> 16);
}

union AB16 { v16bf v; v8bf h[2]; };
static __device__ __forceinline__ v16bf make_frag(v8bf lo, v8bf hi) {
    AB16 u; u.h[0] = lo; u.h[1] = hi; return u.v;
}

#define WMMA_BF16(a, b, c) \
    __builtin_amdgcn_wmma_f32_16x16x32_bf16(false, (a), false, (b), (short)0, (c), false, false)

// ---------------- Stage 1: x fp32 -> bf16 ----------------
__global__ void k_f32_to_bf16(const float* __restrict__ src,
                              unsigned short* __restrict__ dst, int n) {
    int i = blockIdx.x * blockDim.x + threadIdx.x;
    if (i < n) dst[i] = f32_to_bf16(src[i]);
}

// ---------------- Stage 2: transpose weight slab fp32 -> bf16 (N-major) ----
// dst[c*1024 + k] = src[k*srcStride + colOff + c]
__global__ void k_transpose_bf16(const float* __restrict__ src,
                                 unsigned short* __restrict__ dst,
                                 int srcStride, int colOff) {
    __shared__ float tile[32][33];
    const int ct = blockIdx.x * 32, kt = blockIdx.y * 32;
    const int tx = threadIdx.x, ty = threadIdx.y;   // block (32,8)
#pragma unroll
    for (int i = 0; i < 4; ++i)
        tile[ty + 8 * i][tx] =
            src[(size_t)(kt + ty + 8 * i) * srcStride + colOff + ct + tx];
    __syncthreads();
#pragma unroll
    for (int i = 0; i < 4; ++i)
        dst[(size_t)(ct + ty + 8 * i) * 1024 + kt + tx] =
            f32_to_bf16(tile[tx][ty + 8 * i]);
}

// ---------------- Stage 3: KV GEMM ----------------
// Xbf [8192,1024] bf16, Wkvt [2048,1024] bf16 (N-major), out K [B,H,T,64], Vt [B,H,64,T]
__global__ __launch_bounds__(32) void k_gemm_kv(
    const unsigned short* __restrict__ Xbf,
    const unsigned short* __restrict__ Wkvt,
    const float* __restrict__ b_attn,
    unsigned short* __restrict__ Kbuf,
    unsigned short* __restrict__ Vt) {
    const int lane = threadIdx.x;
    const int n = lane & 15, g = lane >> 4;
    const int nb = blockIdx.x * 64;
    const int mb = blockIdx.y * 64;
    v8f acc[4][4] = {};
    for (int kk = 0; kk < 1024; kk += 32) {
        v16bf a[4], b[4];
#pragma unroll
        for (int i = 0; i < 4; ++i) {
            const unsigned short* pa = Xbf + (size_t)(mb + 16 * i + n) * 1024 + kk;
            a[i] = make_frag(*(const v8bf*)(pa + 16 * g),
                             *(const v8bf*)(pa + 16 * g + 8));
        }
#pragma unroll
        for (int j = 0; j < 4; ++j) {
            const unsigned short* pb =
                Wkvt + (size_t)(nb + 16 * j + n) * 1024 + kk + 16 * g;
            b[j] = *(const v16bf*)pb;
        }
#pragma unroll
        for (int i = 0; i < 4; ++i)
#pragma unroll
            for (int j = 0; j < 4; ++j)
                acc[i][j] = WMMA_BF16(a[i], b[j], acc[i][j]);
    }
    float bias[4];
#pragma unroll
    for (int j = 0; j < 4; ++j) bias[j] = b_attn[1024 + nb + 16 * j + n];
#pragma unroll
    for (int i = 0; i < 4; ++i) {
#pragma unroll
        for (int r = 0; r < 8; ++r) {
            const int rowm = mb + 16 * i + r + 8 * g;
            const int bb = rowm >> 11, t = rowm & 2047;
#pragma unroll
            for (int j = 0; j < 4; ++j) {
                const int col = nb + 16 * j + n;
                const float val = acc[i][j][r] + bias[j];
                if (col < 1024) {  // K half
                    const int h = col >> 6, d = col & 63;
                    Kbuf[(((size_t)bb * 16 + h) * 2048 + t) * 64 + d] = f32_to_bf16(val);
                } else {           // V half, stored transposed [hd, T]
                    const int c2 = col - 1024;
                    const int h = c2 >> 6, d = c2 & 63;
                    Vt[(((size_t)bb * 16 + h) * 64 + d) * 2048 + t] = f32_to_bf16(val);
                }
            }
        }
    }
}

// ---------------- Stage 4: causal flash attention (q = k bug) -------------
__global__ __launch_bounds__(32) void k_attn(
    const unsigned short* __restrict__ Kbuf,
    const unsigned short* __restrict__ Vt,
    unsigned short* __restrict__ Aout) {
    __shared__ unsigned short plds[16 * 64];
    const int lane = threadIdx.x;
    const int n = lane & 15, g = lane >> 4;
    const int qb = blockIdx.x * 16;
    const int bh = blockIdx.y;
    const int b = bh >> 4, h = bh & 15;
    const unsigned short* Kb = Kbuf + (size_t)bh * 2048 * 64;
    const unsigned short* Vb = Vt + (size_t)bh * 64 * 2048;

    v16bf qf[2];
#pragma unroll
    for (int j = 0; j < 2; ++j) {   // Q == K rows qb..qb+15 (reference bug)
        const unsigned short* p = Kb + (size_t)(qb + n) * 64 + 32 * j;
        qf[j] = make_frag(*(const v8bf*)(p + 16 * g),
                          *(const v8bf*)(p + 16 * g + 8));
    }
    v8f o[4] = {};
    float mrow[8], lrow[8];
#pragma unroll
    for (int r = 0; r < 8; ++r) { mrow[r] = -__builtin_inff(); lrow[r] = 0.f; }
    const float scale = 0.03125f;  // 1024^-0.5 (reference scales by D, not hd)

    for (int kk = 0; kk < qb + 16; kk += 64) {
        v8f s[4] = {};
#pragma unroll
        for (int j = 0; j < 2; ++j)
#pragma unroll
            for (int t = 0; t < 4; ++t) {
                const unsigned short* p =
                    Kb + (size_t)(kk + 16 * t + n) * 64 + 32 * j + 16 * g;
                const v16bf kf = *(const v16bf*)p;
                s[t] = WMMA_BF16(qf[j], kf, s[t]);
            }
#pragma unroll
        for (int r = 0; r < 8; ++r) {
            const int qrow = qb + r + 8 * g;
            float sv[4];
#pragma unroll
            for (int t = 0; t < 4; ++t) {
                const int kcol = kk + 16 * t + n;
                const float x = s[t][r] * scale;
                sv[t] = (kcol > qrow) ? -__builtin_inff() : x;
            }
            float mx = fmaxf(fmaxf(sv[0], sv[1]), fmaxf(sv[2], sv[3]));
#pragma unroll
            for (int d = 1; d < 16; d <<= 1)
                mx = fmaxf(mx, __shfl_xor(mx, d, 32));
            const float mnew = fmaxf(mrow[r], mx);
            const float corr = __expf(mrow[r] - mnew);
            mrow[r] = mnew;
            float rs = 0.f;
#pragma unroll
            for (int t = 0; t < 4; ++t) {
                const float pv = __expf(sv[t] - mnew);
                rs += pv;
                plds[(r + 8 * g) * 64 + 16 * t + n] = f32_to_bf16(pv);
            }
#pragma unroll
            for (int d = 1; d < 16; d <<= 1)
                rs += __shfl_xor(rs, d, 32);
            lrow[r] = lrow[r] * corr + rs;
#pragma unroll
            for (int dt = 0; dt < 4; ++dt) o[dt][r] *= corr;
        }
        __syncthreads();  // single-wave block: lowers to LDS wait (+NOP barrier)
        v16bf pa[2];
#pragma unroll
        for (int j = 0; j < 2; ++j) {
            const unsigned short* p = plds + n * 64 + 32 * j;
            pa[j] = make_frag(*(const v8bf*)(p + 16 * g),
                              *(const v8bf*)(p + 16 * g + 8));
        }
#pragma unroll
        for (int j = 0; j < 2; ++j)
#pragma unroll
            for (int dt = 0; dt < 4; ++dt) {
                const unsigned short* p =
                    Vb + (size_t)(16 * dt + n) * 2048 + kk + 32 * j + 16 * g;
                const v16bf vf = *(const v16bf*)p;
                o[dt] = WMMA_BF16(pa[j], vf, o[dt]);
            }
        __syncthreads();  // WAR on plds before next chunk
    }
#pragma unroll
    for (int r = 0; r < 8; ++r) {
        const float invl = 1.f / lrow[r];
        const size_t row = (size_t)b * 2048 + qb + r + 8 * g;
#pragma unroll
        for (int dt = 0; dt < 4; ++dt)
            Aout[row * 1024 + h * 64 + 16 * dt + n] =
                f32_to_bf16(o[dt][r] * invl);
    }
}

// ---------------- Stage 5: output projection ----------------
__global__ __launch_bounds__(32) void k_gemm_proj(
    const unsigned short* __restrict__ Abf,
    const unsigned short* __restrict__ Wpt,
    const float* __restrict__ b_proj,
    float* __restrict__ out) {
    const int lane = threadIdx.x;
    const int n = lane & 15, g = lane >> 4;
    const int nb = blockIdx.x * 64;
    const int mb = blockIdx.y * 64;
    v8f acc[4][4] = {};
    for (int kk = 0; kk < 1024; kk += 32) {
        v16bf a[4], b[4];
#pragma unroll
        for (int i = 0; i < 4; ++i) {
            const unsigned short* pa = Abf + (size_t)(mb + 16 * i + n) * 1024 + kk;
            a[i] = make_frag(*(const v8bf*)(pa + 16 * g),
                             *(const v8bf*)(pa + 16 * g + 8));
        }
#pragma unroll
        for (int j = 0; j < 4; ++j) {
            const unsigned short* pb =
                Wpt + (size_t)(nb + 16 * j + n) * 1024 + kk + 16 * g;
            b[j] = *(const v16bf*)pb;
        }
#pragma unroll
        for (int i = 0; i < 4; ++i)
#pragma unroll
            for (int j = 0; j < 4; ++j)
                acc[i][j] = WMMA_BF16(a[i], b[j], acc[i][j]);
    }
    float bias[4];
#pragma unroll
    for (int j = 0; j < 4; ++j) bias[j] = b_proj[nb + 16 * j + n];
#pragma unroll
    for (int i = 0; i < 4; ++i)
#pragma unroll
        for (int r = 0; r < 8; ++r) {
            const size_t rowm = mb + 16 * i + r + 8 * g;
#pragma unroll
            for (int j = 0; j < 4; ++j)
                out[rowm * 1024 + nb + 16 * j + n] = acc[i][j][r] + bias[j];
        }
}

extern "C" void kernel_launch(void* const* d_in, const int* in_sizes, int n_in,
                              void* d_out, int out_size, void* d_ws, size_t ws_size,
                              hipStream_t stream) {
    (void)in_sizes; (void)n_in; (void)out_size; (void)ws_size;
    const float* x      = (const float*)d_in[0];
    const float* W_attn = (const float*)d_in[1];
    const float* b_attn = (const float*)d_in[2];
    const float* W_proj = (const float*)d_in[3];
    const float* b_proj = (const float*)d_in[4];
    float* out = (float*)d_out;

    char* ws = (char*)d_ws;
    size_t off = 0;
    unsigned short* Xbf  = (unsigned short*)(ws + off); off += (size_t)8192 * 1024 * 2;
    unsigned short* Wkvt = (unsigned short*)(ws + off); off += (size_t)2048 * 1024 * 2;
    unsigned short* Wpt  = (unsigned short*)(ws + off); off += (size_t)1024 * 1024 * 2;
    unsigned short* Kbuf = (unsigned short*)(ws + off); off += (size_t)64 * 2048 * 64 * 2;
    unsigned short* Vt   = (unsigned short*)(ws + off); off += (size_t)64 * 2048 * 64 * 2;
    unsigned short* Aout = (unsigned short*)(ws + off);

    k_f32_to_bf16<<<(8388608 + 255) / 256, 256, 0, stream>>>(x, Xbf, 8388608);
    k_transpose_bf16<<<dim3(64, 32), dim3(32, 8), 0, stream>>>(W_attn, Wkvt, 3072, 1024);
    k_transpose_bf16<<<dim3(32, 32), dim3(32, 8), 0, stream>>>(W_proj, Wpt, 1024, 0);
    k_gemm_kv<<<dim3(32, 128), 32, 0, stream>>>(Xbf, Wkvt, b_attn, Kbuf, Vt);
    k_attn<<<dim3(128, 64), 32, 0, stream>>>(Kbuf, Vt, Aout);
    k_gemm_proj<<<dim3(16, 128), 32, 0, stream>>>(Aout, Wpt, b_proj, out);
}
